// MoEFFN_18322330485023
// MI455X (gfx1250) — matrix operand order repaired and checked by reading, not compiled, not run on hardware
//
#include <hip/hip_runtime.h>
#include <stdint.h>

#define NEXP 8
#define TOPK 2
#define HDIM 1024
#define FDIM 4096
#define LN_EPS 1e-12f

#define ASTRIDE 40   // halves per LDS A row: 32 data + 8 pad (80B, 16B aligned)
#define BSTRIDE 40   // halves per LDS B row

typedef __attribute__((ext_vector_type(16))) __bf16 v16bf;
typedef __attribute__((ext_vector_type(8)))  __bf16 v8bf;
typedef __attribute__((ext_vector_type(8)))  float  v8f;
typedef __attribute__((__vector_size__(16))) int    v4i;   // matches async-lds builtin

// ---------------------------------------------------------------------------
// async global->LDS helpers (gfx1250 ASYNCcnt path, with portable fallbacks)
// ---------------------------------------------------------------------------
static __device__ __forceinline__ void async_copy16(const unsigned short* g,
                                                    unsigned short* l) {
#if __has_builtin(__builtin_amdgcn_global_load_async_to_lds_b128)
    __builtin_amdgcn_global_load_async_to_lds_b128(
        (__attribute__((address_space(1))) v4i*)g,
        (__attribute__((address_space(3))) v4i*)l, 0, 0);
#else
    *(v8bf*)l = *(const v8bf*)g;          // regs + ds_store fallback
#endif
}

static __device__ __forceinline__ void wait_async_all() {
#if __has_builtin(__builtin_amdgcn_s_wait_asynccnt)
    __builtin_amdgcn_s_wait_asynccnt(0);
#else
    asm volatile("s_wait_asynccnt 0x0" ::: "memory");
#endif
}

// ---------------------------------------------------------------------------
// fp32 -> bf16 (round-to-nearest-even)
// ---------------------------------------------------------------------------
static __device__ __forceinline__ unsigned short f2bf_rne(float f) {
    union { float f; unsigned u; } v; v.f = f;
    unsigned u = v.u;
    if ((u & 0x7F800000u) == 0x7F800000u) {           // Inf / NaN
        unsigned short r = (unsigned short)(u >> 16);
        if (u & 0x007FFFFFu) r |= 0x0040;             // quiet the NaN
        return r;
    }
    return (unsigned short)((u + 0x7FFFu + ((u >> 16) & 1u)) >> 16);
}

__global__ void convert_bf16_kernel(const float* __restrict__ src,
                                    unsigned short* __restrict__ dst, int n) {
    int i = blockIdx.x * blockDim.x + threadIdx.x;
    int stride = gridDim.x * blockDim.x;
    for (; i < n; i += stride) dst[i] = f2bf_rne(src[i]);
}

// ---------------------------------------------------------------------------
// Router: one wave32 per token. logits = x @ Wr^T + br ; top-2 ; softmax.
// ---------------------------------------------------------------------------
__global__ void router_kernel(const float* __restrict__ x,
                              const float* __restrict__ Wr,
                              const float* __restrict__ br,
                              int* __restrict__ tok_idx,    // [T,2]
                              float* __restrict__ tok_w,    // [T,2]
                              int* __restrict__ counts,     // [E]
                              int T) {
    int wave = (blockIdx.x * blockDim.x + threadIdx.x) >> 5;
    int lane = threadIdx.x & 31;
    if (wave >= T) return;
    const int t = wave;

    float acc[NEXP];
#pragma unroll
    for (int e = 0; e < NEXP; ++e) acc[e] = 0.0f;

    const float* xr = x + (size_t)t * HDIM;
    for (int h = lane; h < HDIM; h += 32) {
        float xv = xr[h];
#pragma unroll
        for (int e = 0; e < NEXP; ++e) acc[e] += xv * Wr[e * HDIM + h];
    }
#pragma unroll
    for (int e = 0; e < NEXP; ++e) {
#pragma unroll
        for (int off = 16; off > 0; off >>= 1)
            acc[e] += __shfl_xor(acc[e], off, 32);
    }

    if (lane == 0) {
        float lg[NEXP];
#pragma unroll
        for (int e = 0; e < NEXP; ++e) lg[e] = acc[e] + br[e];
        int i0 = 0;
#pragma unroll
        for (int e = 1; e < NEXP; ++e) if (lg[e] > lg[i0]) i0 = e;   // ties -> lowest idx
        int i1 = (i0 == 0) ? 1 : 0;
#pragma unroll
        for (int e = 0; e < NEXP; ++e)
            if (e != i0 && lg[e] > lg[i1]) i1 = e;
        float m  = fmaxf(lg[i0], lg[i1]);
        float w0 = __expf(lg[i0] - m);
        float w1 = __expf(lg[i1] - m);
        float s  = w0 + w1;
        tok_idx[t * 2 + 0] = i0;  tok_w[t * 2 + 0] = w0 / s;
        tok_idx[t * 2 + 1] = i1;  tok_w[t * 2 + 1] = w1 / s;
        atomicAdd(&counts[i0], 1);
        atomicAdd(&counts[i1], 1);
    }
}

__global__ void base_kernel(const int* __restrict__ counts, int* __restrict__ basep) {
    if (blockIdx.x == 0 && threadIdx.x == 0) {
        int s = 0;
        for (int e = 0; e < NEXP; ++e) { basep[e] = s; s += counts[e]; }
        basep[NEXP] = s;
    }
}

__global__ void scatter_kernel(const int* __restrict__ tok_idx,
                               const float* __restrict__ tok_w,
                               const int* __restrict__ basep,
                               int* __restrict__ cursor,
                               int* __restrict__ pair_token,
                               float* __restrict__ pair_cw, int T) {
    int t = blockIdx.x * blockDim.x + threadIdx.x;
    if (t >= T) return;
#pragma unroll
    for (int k = 0; k < TOPK; ++k) {
        int e = tok_idx[t * 2 + k];
        int slot = atomicAdd(&cursor[e], 1);
        int p = basep[e] + slot;
        pair_token[p] = t;
        pair_cw[p]    = tok_w[t * 2 + k];
    }
}

// ---------------------------------------------------------------------------
// WMMA helpers (bf16, 16x16x32). ISA 7.12.2 lane layouts.
// ---------------------------------------------------------------------------
static __device__ __forceinline__ v16bf cat8(v8bf lo, v8bf hi) {
    return __builtin_shufflevector(lo, hi, 0,1,2,3,4,5,6,7,8,9,10,11,12,13,14,15);
}
static __device__ __forceinline__ v8f wmma_bf16(v16bf a, v16bf b, v8f c) {
    return __builtin_amdgcn_wmma_f32_16x16x32_bf16(false, a, false, b, (short)0, c, false, false);
}

// ---------------------------------------------------------------------------
// GEMM1 + exact GELU: act[p, f] = gelu(x[tok(p)] . W1[e][f] + b1[e][f])
// Block 256 thr = 8 waves; tile M=128 x N=128 (wave: 32x64 = 2 A-frags x
// 4 B-frags = 8 WMMA / K-step); LDS double-buffered async staging.
// ---------------------------------------------------------------------------
__global__ __launch_bounds__(256, 2)
void expert_gemm1_gelu(const unsigned short* __restrict__ xbf,   // [T,H]
                       const unsigned short* __restrict__ w1bf,  // [E,F,H]
                       const float* __restrict__ b1,             // [E,F]
                       const int* __restrict__ basep,            // [E+1]
                       const int* __restrict__ pair_token,       // [2T]
                       unsigned short* __restrict__ act)         // [2T,F]
{
    __shared__ unsigned short As[2][128 * ASTRIDE];
    __shared__ unsigned short Bs[2][128 * BSTRIDE];

    const int e      = blockIdx.z;
    const int rbase  = basep[e];
    const int n_rows = basep[e + 1] - rbase;
    const int mblock = blockIdx.y * 128;
    if (mblock >= n_rows) return;                 // block-uniform exit
    const int n0   = blockIdx.x * 128;
    const int tid  = threadIdx.x;
    const int wave = tid >> 5, lane = tid & 31;
    const int wm   = (wave & 3) * 32, wn = (wave >> 2) * 64;
    const int r    = lane & 15, hh = lane >> 4;

    // ---- staging: 512 A chunks + 512 B chunks of 16B; 2+2 per thread
    // chunk c: row/col = c>>2, 16B-quarter q = c&3 (row slice is 64B)
    const int q = tid & 3;
    const int a_row0 = tid >> 2, a_row1 = (tid + 256) >> 2;        // 0..63, 64..127
    int a_r0 = mblock + a_row0; if (a_r0 >= n_rows) a_r0 = n_rows - 1;
    int a_r1 = mblock + a_row1; if (a_r1 >= n_rows) a_r1 = n_rows - 1;
    const unsigned short* a_g0 = xbf + (size_t)pair_token[rbase + a_r0] * HDIM + q * 8;
    const unsigned short* a_g1 = xbf + (size_t)pair_token[rbase + a_r1] * HDIM + q * 8;
    unsigned short* a_l0[2] = { &As[0][a_row0 * ASTRIDE + q * 8],
                                &As[1][a_row0 * ASTRIDE + q * 8] };
    unsigned short* a_l1[2] = { &As[0][a_row1 * ASTRIDE + q * 8],
                                &As[1][a_row1 * ASTRIDE + q * 8] };

    const int b_col0 = tid >> 2, b_col1 = (tid + 256) >> 2;
    const unsigned short* b_g0 = w1bf + ((size_t)e * FDIM + n0 + b_col0) * HDIM + q * 8;
    const unsigned short* b_g1 = w1bf + ((size_t)e * FDIM + n0 + b_col1) * HDIM + q * 8;
    unsigned short* b_l0[2] = { &Bs[0][b_col0 * BSTRIDE + q * 8],
                                &Bs[1][b_col0 * BSTRIDE + q * 8] };
    unsigned short* b_l1[2] = { &Bs[0][b_col1 * BSTRIDE + q * 8],
                                &Bs[1][b_col1 * BSTRIDE + q * 8] };

    auto stage = [&](int k0, int buf) {
        async_copy16(a_g0 + k0, a_l0[buf]);
        async_copy16(a_g1 + k0, a_l1[buf]);
        async_copy16(b_g0 + k0, b_l0[buf]);
        async_copy16(b_g1 + k0, b_l1[buf]);
    };

    stage(0, 0);
    wait_async_all();
    __syncthreads();

    v8f c00 = {}, c01 = {}, c02 = {}, c03 = {};
    v8f c10 = {}, c11 = {}, c12 = {}, c13 = {};
    int buf = 0;
    for (int k0 = 0; k0 < HDIM; k0 += 32) {
        if (k0 + 32 < HDIM) stage(k0 + 32, buf ^ 1);   // overlap with compute

        const unsigned short* Ab = As[buf];
        const unsigned short* Bb = Bs[buf];
        v16bf a0 = cat8(*(const v8bf*)&Ab[(wm +      r) * ASTRIDE + hh * 16],
                        *(const v8bf*)&Ab[(wm +      r) * ASTRIDE + hh * 16 + 8]);
        v16bf a1 = cat8(*(const v8bf*)&Ab[(wm + 16 + r) * ASTRIDE + hh * 16],
                        *(const v8bf*)&Ab[(wm + 16 + r) * ASTRIDE + hh * 16 + 8]);
        v16bf b0 = *(const v16bf*)&Bb[(wn +  0 + r) * BSTRIDE + hh * 16];
        v16bf bv1= *(const v16bf*)&Bb[(wn + 16 + r) * BSTRIDE + hh * 16];
        v16bf bv2= *(const v16bf*)&Bb[(wn + 32 + r) * BSTRIDE + hh * 16];
        v16bf bv3= *(const v16bf*)&Bb[(wn + 48 + r) * BSTRIDE + hh * 16];
        c00 = wmma_bf16(a0, b0,  c00);
        c01 = wmma_bf16(a0, bv1, c01);
        c02 = wmma_bf16(a0, bv2, c02);
        c03 = wmma_bf16(a0, bv3, c03);
        c10 = wmma_bf16(a1, b0,  c10);
        c11 = wmma_bf16(a1, bv1, c11);
        c12 = wmma_bf16(a1, bv2, c12);
        c13 = wmma_bf16(a1, bv3, c13);

        wait_async_all();
        __syncthreads();
        buf ^= 1;
    }

    // C layout: elem i -> row = wm + m*16 + hh*8 + i, col = n0+wn+s*16+r
#pragma unroll
    for (int m = 0; m < 2; ++m) {
#pragma unroll
        for (int i = 0; i < 8; ++i) {
            int orow = mblock + wm + m * 16 + hh * 8 + i;
            if (orow >= n_rows) continue;
            size_t rowoff = (size_t)(rbase + orow) * FDIM;
            float vals[4] = { m ? c10[i] : c00[i], m ? c11[i] : c01[i],
                              m ? c12[i] : c02[i], m ? c13[i] : c03[i] };
#pragma unroll
            for (int s = 0; s < 4; ++s) {
                int col = n0 + wn + s * 16 + r;
                float v = vals[s] + b1[e * FDIM + col];
                v = 0.5f * v * (1.0f + erff(v * 0.70710678118654752f));  // exact gelu
                act[rowoff + col] = f2bf_rne(v);
            }
        }
    }
}

// ---------------------------------------------------------------------------
// GEMM2 + scaled scatter: acc[tok] += cw * (act[p] . W2[e][h] + b2[e][h])
// ---------------------------------------------------------------------------
__global__ __launch_bounds__(256, 2)
void expert_gemm2_scatter(const unsigned short* __restrict__ act,   // [2T,F]
                          const unsigned short* __restrict__ w2bf,  // [E,H,F]
                          const float* __restrict__ b2,             // [E,H]
                          const int* __restrict__ basep,
                          const int* __restrict__ pair_token,
                          const float* __restrict__ pair_cw,
                          float* __restrict__ acc)                  // [T,H]
{
    __shared__ unsigned short As[2][128 * ASTRIDE];
    __shared__ unsigned short Bs[2][128 * BSTRIDE];

    const int e      = blockIdx.z;
    const int rbase  = basep[e];
    const int n_rows = basep[e + 1] - rbase;
    const int mblock = blockIdx.y * 128;
    if (mblock >= n_rows) return;
    const int n0   = blockIdx.x * 128;
    const int tid  = threadIdx.x;
    const int wave = tid >> 5, lane = tid & 31;
    const int wm   = (wave & 3) * 32, wn = (wave >> 2) * 64;
    const int r    = lane & 15, hh = lane >> 4;

    const int q = tid & 3;
    const int a_row0 = tid >> 2, a_row1 = (tid + 256) >> 2;
    int a_r0 = mblock + a_row0; if (a_r0 >= n_rows) a_r0 = n_rows - 1;
    int a_r1 = mblock + a_row1; if (a_r1 >= n_rows) a_r1 = n_rows - 1;
    const unsigned short* a_g0 = act + (size_t)(rbase + a_r0) * FDIM + q * 8;
    const unsigned short* a_g1 = act + (size_t)(rbase + a_r1) * FDIM + q * 8;
    unsigned short* a_l0[2] = { &As[0][a_row0 * ASTRIDE + q * 8],
                                &As[1][a_row0 * ASTRIDE + q * 8] };
    unsigned short* a_l1[2] = { &As[0][a_row1 * ASTRIDE + q * 8],
                                &As[1][a_row1 * ASTRIDE + q * 8] };

    const int b_col0 = tid >> 2, b_col1 = (tid + 256) >> 2;
    const unsigned short* b_g0 = w2bf + ((size_t)e * HDIM + n0 + b_col0) * FDIM + q * 8;
    const unsigned short* b_g1 = w2bf + ((size_t)e * HDIM + n0 + b_col1) * FDIM + q * 8;
    unsigned short* b_l0[2] = { &Bs[0][b_col0 * BSTRIDE + q * 8],
                                &Bs[1][b_col0 * BSTRIDE + q * 8] };
    unsigned short* b_l1[2] = { &Bs[0][b_col1 * BSTRIDE + q * 8],
                                &Bs[1][b_col1 * BSTRIDE + q * 8] };

    auto stage = [&](int k0, int buf) {
        async_copy16(a_g0 + k0, a_l0[buf]);
        async_copy16(a_g1 + k0, a_l1[buf]);
        async_copy16(b_g0 + k0, b_l0[buf]);
        async_copy16(b_g1 + k0, b_l1[buf]);
    };

    stage(0, 0);
    wait_async_all();
    __syncthreads();

    v8f c00 = {}, c01 = {}, c02 = {}, c03 = {};
    v8f c10 = {}, c11 = {}, c12 = {}, c13 = {};
    int buf = 0;
    for (int k0 = 0; k0 < FDIM; k0 += 32) {
        if (k0 + 32 < FDIM) stage(k0 + 32, buf ^ 1);

        const unsigned short* Ab = As[buf];
        const unsigned short* Bb = Bs[buf];
        v16bf a0 = cat8(*(const v8bf*)&Ab[(wm +      r) * ASTRIDE + hh * 16],
                        *(const v8bf*)&Ab[(wm +      r) * ASTRIDE + hh * 16 + 8]);
        v16bf a1 = cat8(*(const v8bf*)&Ab[(wm + 16 + r) * ASTRIDE + hh * 16],
                        *(const v8bf*)&Ab[(wm + 16 + r) * ASTRIDE + hh * 16 + 8]);
        v16bf b0 = *(const v16bf*)&Bb[(wn +  0 + r) * BSTRIDE + hh * 16];
        v16bf bv1= *(const v16bf*)&Bb[(wn + 16 + r) * BSTRIDE + hh * 16];
        v16bf bv2= *(const v16bf*)&Bb[(wn + 32 + r) * BSTRIDE + hh * 16];
        v16bf bv3= *(const v16bf*)&Bb[(wn + 48 + r) * BSTRIDE + hh * 16];
        c00 = wmma_bf16(a0, b0,  c00);
        c01 = wmma_bf16(a0, bv1, c01);
        c02 = wmma_bf16(a0, bv2, c02);
        c03 = wmma_bf16(a0, bv3, c03);
        c10 = wmma_bf16(a1, b0,  c10);
        c11 = wmma_bf16(a1, bv1, c11);
        c12 = wmma_bf16(a1, bv2, c12);
        c13 = wmma_bf16(a1, bv3, c13);

        wait_async_all();
        __syncthreads();
        buf ^= 1;
    }

#pragma unroll
    for (int m = 0; m < 2; ++m) {
#pragma unroll
        for (int i = 0; i < 8; ++i) {
            int orow = mblock + wm + m * 16 + hh * 8 + i;
            if (orow >= n_rows) continue;
            int   tok = pair_token[rbase + orow];
            float cw  = pair_cw[rbase + orow];
            float vals[4] = { m ? c10[i] : c00[i], m ? c11[i] : c01[i],
                              m ? c12[i] : c02[i], m ? c13[i] : c03[i] };
#pragma unroll
            for (int s = 0; s < 4; ++s) {
                int col = n0 + wn + s * 16 + r;
                float v = vals[s] + b2[e * HDIM + col];
                atomicAdd(&acc[(size_t)tok * HDIM + col], cw * v);  // 2 adds/elt: deterministic
            }
        }
    }
}

// ---------------------------------------------------------------------------
// Residual + LayerNorm (one 256-thread block per token, H=1024)
// ---------------------------------------------------------------------------
__global__ void ln_kernel(const float* __restrict__ acc,
                          const float* __restrict__ x,
                          const float* __restrict__ lnw,
                          const float* __restrict__ lnb,
                          float* __restrict__ out) {
    const int t = blockIdx.x;
    const int tid = threadIdx.x;
    __shared__ float red_s[8], red_q[8];

    const float* ar = acc + (size_t)t * HDIM;
    const float* xr = x   + (size_t)t * HDIM;
    float zv[4], sum = 0.0f, sq = 0.0f;
#pragma unroll
    for (int j = 0; j < 4; ++j) {
        int h = tid + j * 256;
        float z = ar[h] + xr[h];
        zv[j] = z; sum += z; sq += z * z;
    }
#pragma unroll
    for (int off = 16; off > 0; off >>= 1) {
        sum += __shfl_xor(sum, off, 32);
        sq  += __shfl_xor(sq,  off, 32);
    }
    int wave = tid >> 5;
    if ((tid & 31) == 0) { red_s[wave] = sum; red_q[wave] = sq; }
    __syncthreads();
    float ts = 0.0f, tq = 0.0f;
#pragma unroll
    for (int w = 0; w < 8; ++w) { ts += red_s[w]; tq += red_q[w]; }
    float mu  = ts * (1.0f / HDIM);
    float var = tq * (1.0f / HDIM) - mu * mu;
    float inv = rsqrtf(var + LN_EPS);
#pragma unroll
    for (int j = 0; j < 4; ++j) {
        int h = tid + j * 256;
        out[(size_t)t * HDIM + h] = (zv[j] - mu) * inv * lnw[h] + lnb[h];
    }
}

// ---------------------------------------------------------------------------
// Launcher
// ---------------------------------------------------------------------------
extern "C" void kernel_launch(void* const* d_in, const int* in_sizes, int n_in,
                              void* d_out, int out_size, void* d_ws, size_t ws_size,
                              hipStream_t stream) {
    const float* x   = (const float*)d_in[0];
    const float* Wr  = (const float*)d_in[1];
    const float* br  = (const float*)d_in[2];
    const float* W1  = (const float*)d_in[3];
    const float* b1  = (const float*)d_in[4];
    const float* W2  = (const float*)d_in[5];
    const float* b2  = (const float*)d_in[6];
    const float* lnw = (const float*)d_in[7];
    const float* lnb = (const float*)d_in[8];
    float* out = (float*)d_out;

    const int T = in_sizes[0] / HDIM;     // 8192 tokens
    const int P = TOPK * T;               // 16384 routed pairs

    char* ws = (char*)d_ws;
    size_t off = 0;
    auto take = [&](size_t bytes) -> char* {
        char* p = ws + off;
        off = (off + bytes + 255) & ~(size_t)255;
        return p;
    };
    unsigned short* w1bf = (unsigned short*)take((size_t)NEXP * FDIM * HDIM * 2);
    unsigned short* w2bf = (unsigned short*)take((size_t)NEXP * HDIM * FDIM * 2);
    unsigned short* xbf  = (unsigned short*)take((size_t)T * HDIM * 2);
    unsigned short* act  = (unsigned short*)take((size_t)P * FDIM * 2);
    float* acc      = (float*)take((size_t)T * HDIM * 4);
    int*   counts   = (int*)  take(NEXP * 4);
    int*   cursor   = (int*)  take(NEXP * 4);
    int*   basep    = (int*)  take((NEXP + 1) * 4);
    int*   tok_idx  = (int*)  take((size_t)T * 2 * 4);
    float* tok_w    = (float*)take((size_t)T * 2 * 4);
    int*   pair_tok = (int*)  take((size_t)P * 4);
    float* pair_cw  = (float*)take((size_t)P * 4);

    (void)hipMemsetAsync(acc,    0, (size_t)T * HDIM * 4, stream);
    (void)hipMemsetAsync(counts, 0, NEXP * 4, stream);
    (void)hipMemsetAsync(cursor, 0, NEXP * 4, stream);

    const int nW = NEXP * FDIM * HDIM;
    convert_bf16_kernel<<<2048, 256, 0, stream>>>(W1, w1bf, nW);
    convert_bf16_kernel<<<2048, 256, 0, stream>>>(W2, w2bf, nW);
    convert_bf16_kernel<<<512, 256, 0, stream>>>(x, xbf, T * HDIM);

    router_kernel<<<(T + 7) / 8, 256, 0, stream>>>(x, Wr, br, tok_idx, tok_w, counts, T);
    base_kernel<<<1, 1, 0, stream>>>(counts, basep);
    scatter_kernel<<<(T + 255) / 256, 256, 0, stream>>>(tok_idx, tok_w, basep, cursor,
                                                        pair_tok, pair_cw, T);

    dim3 g1(FDIM / 128, (T + 127) / 128, NEXP);   // padded grid, early-exit on n_rows
    expert_gemm1_gelu<<<g1, 256, 0, stream>>>(xbf, w1bf, b1, basep, pair_tok, act);

    dim3 g2(HDIM / 128, (T + 127) / 128, NEXP);
    expert_gemm2_scatter<<<g2, 256, 0, stream>>>(act, w2bf, b2, basep, pair_tok, pair_cw, acc);

    ln_kernel<<<T, 256, 0, stream>>>(acc, x, lnw, lnb, out);
}
